// drebin_14654428414263
// MI455X (gfx1250) — compile-verified, matrix operand order checked by reading, not run on hardware
//
#include <hip/hip_runtime.h>
#include <stdint.h>

#define NQ   9
#define DIM  512
#define NL   10
#define NCLS 10
#define BATCH 16384
#define RPAD 520          // LDS row stride in f16 (512 + 8 pad -> 1040B, breaks bank conflicts)
#define ROWS_PER_BLK 64   // 4 waves x 16 batch rows

typedef __attribute__((ext_vector_type(16))) _Float16 v16h;
typedef __attribute__((ext_vector_type(8)))  float    v8f;

// ---------------------------------------------------------------------------
// Kernel 1: build the full circuit unitary U (512x512 complex).
// Columns of U are independent: one block per column, column state in LDS,
// 180 gates applied sequentially with workgroup barriers.
// Output: Wr/Wi row-major f16 [s][k] (s = output state index, k = input basis).
// ---------------------------------------------------------------------------
extern "C" __global__ void __launch_bounds__(256)
qc_build_u(const float* __restrict__ rot, const float* __restrict__ crx,
           _Float16* __restrict__ Wr, _Float16* __restrict__ Wi)
{
  __shared__ float cr[DIM], ci[DIM];
  const int col = blockIdx.x;
  const int t   = threadIdx.x;
  for (int s = t; s < DIM; s += 256) { cr[s] = (s == col) ? 1.f : 0.f; ci[s] = 0.f; }
  __syncthreads();

  for (int n = 0; n < NL; ++n) {
    // ---- 9 single-qubit Rot gates: RZ(om) RY(th) RZ(phi)
    for (int q = 0; q < NQ; ++q) {
      const float* rp = rot + (n * NQ + q) * 3;
      float phi = rp[0], th = rp[1], om = rp[2];
      float ct = cosf(0.5f * th), st = sinf(0.5f * th);
      float a  = 0.5f * (phi + om), b = 0.5f * (phi - om);
      float ca = cosf(a), sa = sinf(a), cb = cosf(b), sb = sinf(b);
      float u00r =  ca * ct, u00i = -sa * ct;
      float u01r = -cb * st, u01i = -sb * st;
      float u10r =  cb * st, u10i = -sb * st;
      float u11r =  ca * ct, u11i =  sa * ct;
      int bit = 8 - q;                      // qubit 0 = MSB of the 9-bit index
      int stride = 1 << bit;
      int i0 = ((t >> bit) << (bit + 1)) | (t & (stride - 1));
      int i1 = i0 | stride;
      float v0r = cr[i0], v0i = ci[i0], v1r = cr[i1], v1i = ci[i1];
      float w0r = u00r*v0r - u00i*v0i + u01r*v1r - u01i*v1i;
      float w0i = u00r*v0i + u00i*v0r + u01r*v1i + u01i*v1r;
      float w1r = u10r*v0r - u10i*v0i + u11r*v1r - u11i*v1i;
      float w1i = u10r*v0i + u10i*v0r + u11r*v1i + u11i*v1r;
      cr[i0] = w0r; ci[i0] = w0i; cr[i1] = w1r; ci[i1] = w1i;
      __syncthreads();
    }
    // ---- 9 CRX gates: control q, target (q+1)%9; acts on control==1 subspace
    for (int q = 0; q < NQ; ++q) {
      float tt = crx[n * NQ + q];
      float c_ = cosf(0.5f * tt), s_ = sinf(0.5f * tt);
      if (t < 128) {
        int bctl = 8 - q;
        int btgt = 8 - ((q + 1) % NQ);
        int lo = bctl < btgt ? bctl : btgt;
        int hi = bctl < btgt ? btgt : bctl;
        int xidx = t;
        xidx = ((xidx >> lo) << (lo + 1)) | (xidx & ((1 << lo) - 1));
        xidx = ((xidx >> hi) << (hi + 1)) | (xidx & ((1 << hi) - 1));
        int i0 = xidx | (1 << bctl);        // control=1, target=0
        int i1 = i0 | (1 << btgt);          // control=1, target=1
        float v0r = cr[i0], v0i = ci[i0], v1r = cr[i1], v1i = ci[i1];
        // RX: w0 = c*v0 - i*s*v1 ; w1 = -i*s*v0 + c*v1
        float w0r = c_*v0r + s_*v1i;
        float w0i = c_*v0i - s_*v1r;
        float w1r = c_*v1r + s_*v0i;
        float w1i = c_*v1i - s_*v0r;
        cr[i0] = w0r; ci[i0] = w0i; cr[i1] = w1r; ci[i1] = w1i;
      }
      __syncthreads();
    }
  }
  for (int s = t; s < DIM; s += 256) {
    Wr[(size_t)s * DIM + col] = (_Float16)cr[s];
    Wi[(size_t)s * DIM + col] = (_Float16)ci[s];
  }
}

// ---------------------------------------------------------------------------
// Async global->LDS copy (CDNA5 VGLOBAL async op, tracked by ASYNCcnt)
// ---------------------------------------------------------------------------
__device__ __forceinline__ void async_b128(unsigned ldsaddr, const void* gp) {
  unsigned long long ga = (unsigned long long)(uintptr_t)gp;
  asm volatile("global_load_async_to_lds_b128 %0, %1, off"
               :: "v"(ldsaddr), "v"(ga) : "memory");
}
__device__ __forceinline__ void wait_async_le16() {
  asm volatile("s_wait_asynccnt 0x10" ::: "memory");
}
__device__ __forceinline__ void wait_async_0() {
  asm volatile("s_wait_asynccnt 0x0" ::: "memory");
}

// Load one v16h WMMA operand fragment: two contiguous 16B runs from an LDS row.
__device__ __forceinline__ v16h frag_ld(const _Float16* rowp, int off_lo) {
  union { v16h v; uint4 u[2]; } r;
  r.u[0] = *(const uint4*)(rowp + off_lo);        // K = off_lo .. +7
  r.u[1] = *(const uint4*)(rowp + off_lo + 16);   // K = off_lo+16 .. +23
  return r.v;
}

// ---------------------------------------------------------------------------
// Kernel 2: fused  x-normalize -> complex matmul (WMMA f16) -> probs ->
//           <Z_q> (parity-split accumulation) -> logits -> log_softmax.
// Block: 128 threads (4 waves), 64 batch rows; wave w owns rows w*16..w*16+15.
// A fragments live in 128 VGPRs for the whole kernel; s-tiles of U streamed
// through double-buffered LDS via async copies -> 2 ds_load_b128 per WMMA.
// ---------------------------------------------------------------------------
extern "C" __global__ void __launch_bounds__(128)
qc_main(const float* __restrict__ x, const float* __restrict__ fc_w,
        const float* __restrict__ fc_b,
        const _Float16* __restrict__ Wr, const _Float16* __restrict__ Wi,
        float* __restrict__ out)
{
  __shared__ _Float16 Xh[ROWS_PER_BLK][RPAD];   // normalized x, f16 (wave-local use)
  __shared__ _Float16 Ut[2][2][16][RPAD];       // [buf][re/im][row s][k]
  __shared__ float    Zl[ROWS_PER_BLK][NQ];     // staged <Z_q>

  const int t  = threadIdx.x;
  const int w  = t >> 5;
  const int L  = t & 31;
  const int ln = L & 15;      // N index of WMMA fragments
  const int hf = L >> 4;      // half-wave select
  const int row0 = blockIdx.x * ROWS_PER_BLK;

  // ---- phase 1: per-row L2 normalize -> f16 LDS (coalesced loads, wave-reduced norm)
  for (int rr = 0; rr < 16; ++rr) {
    int r = w * 16 + rr;
    const float* xp = x + (size_t)(row0 + r) * DIM;
    float xv[16]; float ss = 0.f;
#pragma unroll
    for (int j = 0; j < 16; ++j) { xv[j] = xp[j * 32 + L]; ss += xv[j] * xv[j]; }
#pragma unroll
    for (int m = 16; m >= 1; m >>= 1) ss += __shfl_xor(ss, m, 32);
    float rinv = rsqrtf(ss);
    rinv = rinv * (1.5f - 0.5f * ss * rinv * rinv);   // Newton refine
#pragma unroll
    for (int j = 0; j < 16; ++j) Xh[r][j * 32 + L] = (_Float16)(xv[j] * rinv);
  }

  // ---- prefetch s-tile 0 (16 rows of Ur and Ui) via async copies
  {
    _Float16* ub = &Ut[0][0][0][0];
#pragma unroll
    for (int i = 0; i < 8; ++i) {
      int ch = t + i * 128;                 // 1024 chunks of 16B per matrix
      int rowu = ch >> 6, kc = (ch & 63) * 8;
      async_b128((unsigned)(uintptr_t)(ub + rowu * RPAD + kc),
                 Wr + (size_t)rowu * DIM + kc);
      async_b128((unsigned)(uintptr_t)(ub + 16 * RPAD + rowu * RPAD + kc),
                 Wi + (size_t)rowu * DIM + kc);
    }
  }

  // ---- hoist this wave's A fragments into registers (wave-local LDS, in-order DS)
  const int boff = hf * 8;                       // K sub-pattern base per half-wave
  const _Float16* arow = &Xh[w * 16 + ln][0];    // A: row M = lane%16
  v16h aA[16];
#pragma unroll
  for (int kt = 0; kt < 16; ++kt) aA[kt] = frag_ld(arow, kt * 32 + boff);

  // parity-split accumulators: s = st*16 + ln
  //   qubits 0..4: sign depends on st only (wave-uniform)  -> zacc[5][8]
  //   qubits 5..8: sign depends on ln only (fixed per lane) -> +-psum applied at end
  float psum[8];
  float zacc[5][8];
#pragma unroll
  for (int v = 0; v < 8; ++v) {
    psum[v] = 0.f;
#pragma unroll
    for (int q = 0; q < 5; ++q) zacc[q][v] = 0.f;
  }

  for (int st = 0; st < 32; ++st) {
    // prefetch next tile into the other buffer
    if (st + 1 < 32) {
      _Float16* ub = &Ut[(st + 1) & 1][0][0][0];
      int s0n = (st + 1) * 16;
#pragma unroll
      for (int i = 0; i < 8; ++i) {
        int ch = t + i * 128;
        int rowu = ch >> 6, kc = (ch & 63) * 8;
        async_b128((unsigned)(uintptr_t)(ub + rowu * RPAD + kc),
                   Wr + (size_t)(s0n + rowu) * DIM + kc);
        async_b128((unsigned)(uintptr_t)(ub + 16 * RPAD + rowu * RPAD + kc),
                   Wi + (size_t)(s0n + rowu) * DIM + kc);
      }
      wait_async_le16();     // current tile (issued earlier) is complete
    } else {
      wait_async_0();
    }
    __syncthreads();         // tile st visible to all waves

    const int cb = st & 1;
    const _Float16* brow  = &Ut[cb][0][ln][0];   // B real: column N = lane%16 = U row
    const _Float16* birow = &Ut[cb][1][ln][0];   // B imag
    v8f pr = {0.f, 0.f, 0.f, 0.f, 0.f, 0.f, 0.f, 0.f};
    v8f pi = {0.f, 0.f, 0.f, 0.f, 0.f, 0.f, 0.f, 0.f};
#pragma unroll
    for (int kt = 0; kt < 16; ++kt) {
      int off = kt * 32 + boff;
      v16h br = frag_ld(brow, off);
      v16h bi = frag_ld(birow, off);
      pr = __builtin_amdgcn_wmma_f32_16x16x32_f16(false, aA[kt], false, br, (short)0, pr, false, false);
      pi = __builtin_amdgcn_wmma_f32_16x16x32_f16(false, aA[kt], false, bi, (short)0, pi, false, false);
    }
    // epilogue: probs at (m = 8*hf+v, s = st*16+ln)
    float sg[5];
#pragma unroll
    for (int q = 0; q < 5; ++q) sg[q] = ((st >> (4 - q)) & 1) ? -1.f : 1.f;
#pragma unroll
    for (int v = 0; v < 8; ++v) {
      float p = pr[v] * pr[v] + pi[v] * pi[v];
      psum[v] += p;
#pragma unroll
      for (int q = 0; q < 5; ++q) zacc[q][v] = fmaf(sg[q], p, zacc[q][v]);
    }
    __syncthreads();         // all waves done reading buf[st&1] before it is reloaded
  }

  // ---- expand to full z[9][8], then reduce over the 16 s-phases (lanes)
  float z[NQ * 8];
#pragma unroll
  for (int q = 0; q < 5; ++q) {
#pragma unroll
    for (int v = 0; v < 8; ++v) z[q * 8 + v] = zacc[q][v];
  }
#pragma unroll
  for (int q = 5; q < NQ; ++q) {
    float sl = ((ln >> (8 - q)) & 1) ? -1.f : 1.f;
#pragma unroll
    for (int v = 0; v < 8; ++v) z[q * 8 + v] = sl * psum[v];
  }
#pragma unroll
  for (int m = 8; m >= 1; m >>= 1) {
#pragma unroll
    for (int i = 0; i < NQ * 8; ++i) z[i] += __shfl_xor(z[i], m, 32);
  }
  if (ln == 0) {
#pragma unroll
    for (int v = 0; v < 8; ++v) {
#pragma unroll
      for (int q = 0; q < NQ; ++q)
        Zl[w * 16 + hf * 8 + v][q] = z[q * 8 + v];
    }
  }
  __syncthreads();

  // ---- linear layer + log_softmax, one row per thread
  if (t < ROWS_PER_BLK) {
    float zz[NQ];
#pragma unroll
    for (int q = 0; q < NQ; ++q) zz[q] = Zl[t][q];
    float l[NCLS]; float mx = -1e30f;
#pragma unroll
    for (int c = 0; c < NCLS; ++c) {
      float acc = fc_b[c];
#pragma unroll
      for (int q = 0; q < NQ; ++q) acc = fmaf(fc_w[c * NQ + q], zz[q], acc);
      l[c] = acc; mx = fmaxf(mx, acc);
    }
    float sum = 0.f;
#pragma unroll
    for (int c = 0; c < NCLS; ++c) sum += expf(l[c] - mx);
    float ls = mx + logf(sum);
    float* op = out + (size_t)(row0 + t) * NCLS;
#pragma unroll
    for (int c = 0; c < NCLS; ++c) op[c] = l[c] - ls;
  }
}

// ---------------------------------------------------------------------------
extern "C" void kernel_launch(void* const* d_in, const int* in_sizes, int n_in,
                              void* d_out, int out_size, void* d_ws, size_t ws_size,
                              hipStream_t stream) {
  const float* x    = (const float*)d_in[0];
  const float* rot  = (const float*)d_in[1];
  const float* crx  = (const float*)d_in[2];
  const float* fc_w = (const float*)d_in[3];
  const float* fc_b = (const float*)d_in[4];
  float* out = (float*)d_out;

  _Float16* Wr = (_Float16*)d_ws;                 // 512KB
  _Float16* Wi = Wr + (size_t)DIM * DIM;          // 512KB

  qc_build_u<<<DIM, 256, 0, stream>>>(rot, crx, Wr, Wi);
  qc_main<<<BATCH / ROWS_PER_BLK, 128, 0, stream>>>(x, fc_w, fc_b, Wr, Wi, out);
}
